// MultiHeadAttention_33234456936896
// MI455X (gfx1250) — matrix operand order repaired and checked by reading, not compiled, not run on hardware
//
#include <hip/hip_runtime.h>
#include <hip/hip_bf16.h>

#define NUM_HEADS 32
#define HIDDEN    2048
#define HEAD_DIM  64
#define BATCH     2
#define SEQ       2048
#define MROWS     (BATCH * SEQ)   // 4096

// 1/sqrt(HEAD_DIM) * log2(e): folded into Q so softmax runs in exp2 domain.
#define QSCALE 0.18033688011112042f

typedef __attribute__((ext_vector_type(16))) _Float16 v16h;
typedef __attribute__((ext_vector_type(8)))  float    v8f;

union Frag16 { v16h v; unsigned u[8]; _Float16 h[16]; };

// ---------------------------------------------------------------------------
// CDNA5 async global->LDS copy (ASYNCcnt path, no VGPR round-trip).
// VDST operand is the wave-relative LDS byte address = low 32 bits of the
// generic LDS pointer (aperture mapping, ISA 10.2). GV mode: 64-bit vaddr, off.
// ---------------------------------------------------------------------------
static __device__ __forceinline__ void async_b128(void* lds, const void* gptr) {
  unsigned l = (unsigned)(unsigned long long)lds;
  asm volatile("global_load_async_to_lds_b128 %0, %1, off"
               :: "v"(l), "v"(gptr) : "memory");
}
static __device__ __forceinline__ void wait_async0() {
  asm volatile("s_wait_asynccnt 0x0" ::: "memory");
}

// ---------------------------------------------------------------------------
// WMMA fragment loaders (contiguous k = (lane/16)*16 + s in both A and B, 16-bit, wave32).
// Both loaders read 2 x 16B contiguous per lane -> 2 x ds_load_b128 each.
// ---------------------------------------------------------------------------

// A fragment 16x32 f16 from row-major [m][k]: lane holds row m = row+(lane&15);
// K index per VGPR v: k = (lane/16)*16 + v*2  (pairs packed)
static __device__ __forceinline__ v16h load_a_frag(const _Float16* p, int row,
                                                   int stride, int kk, int lane) {
  Frag16 f;
  const unsigned* rp = (const unsigned*)(p + (row + (lane & 15)) * stride + kk);
  const int half = lane >> 4;
#pragma unroll
  for (int v = 0; v < 8; ++v) {
    int k = (half << 4) + (v << 1);
    f.u[v] = rp[k >> 1];
  }
  return f.v;
}

// B fragment where logical B = R^T with R row-major [n][k]
// (K-matrix for Q@K^T, transposed weights, transposed V):
// lane's 16 k-values are contiguous -> pure dword loads.
static __device__ __forceinline__ v16h load_bT_frag(const _Float16* p, int colbase,
                                                    int stride, int kk, int lane) {
  Frag16 f;
  const unsigned* rp = (const unsigned*)(p + (colbase + (lane & 15)) * stride + kk);
  const int kb2 = (lane >> 4) << 3;   // (half*16)/2 dwords
#pragma unroll
  for (int v = 0; v < 8; ++v) f.u[v] = rp[kb2 + v];
  return f.v;
}

// ---------------------------------------------------------------------------
// fp32 -> f16 conversion (plain) and transpose variant for weights.
// ---------------------------------------------------------------------------
__global__ void cvt_f32_f16(const float* __restrict__ in, _Float16* __restrict__ out, int n) {
  int i = blockIdx.x * blockDim.x + threadIdx.x;
  if (i < n) out[i] = (_Float16)in[i];
}

// in: [R][C] fp32 row-major;  out: [C][R] f16 (out[n][k] = in[k][n]).
// Coalesced writes; strided reads stay resident in the 192MB L2.
__global__ void cvt_f32_f16_T(const float* __restrict__ in, _Float16* __restrict__ out,
                              int R, int C) {
  int i = blockIdx.x * blockDim.x + threadIdx.x;
  if (i < R * C) {
    int k = i & (R - 1);          // fast index -> coalesced write
    int n = i >> 11;              // R = 2048
    out[i] = (_Float16)in[(size_t)k * C + n];
  }
}

// ---------------------------------------------------------------------------
// Tiled WMMA GEMM: C[M,N] = A[M,K] @ Bt^T + bias, A row-major [M][K],
// Bt row-major [N][K] (pre-transposed weights). Block tile 128x128,
// K-stage 64, double-buffered async global->LDS staging. 8 waves (2x4),
// each wave 4x2 16x16 tiles.
// MODE 0: fp32 row-major. MODE 1: f16 [B,H,S,d] (*oscale). MODE 2: f16 [B,H,d,S].
// ---------------------------------------------------------------------------
template <int MODE>
__global__ __launch_bounds__(256) void gemm_f16(const _Float16* __restrict__ A,
                                                const _Float16* __restrict__ Bt,
                                                const float* __restrict__ bias,
                                                float* __restrict__ outf,
                                                _Float16* __restrict__ outh,
                                                int M, int N, int K, float oscale) {
  __shared__ _Float16 As[2][128][72];    // 2 x 18.0 KB
  __shared__ _Float16 Bs[2][128][72];    // 2 x 18.0 KB  (rows = n, cols = k)
  const int t = threadIdx.x;
  const int lane = t & 31, wid = t >> 5;
  const int wm = wid >> 2, wn = wid & 3;            // 2 (M) x 4 (N) waves
  const int m0 = blockIdx.y * 128, n0 = blockIdx.x * 128;

  auto stage = [&](int buf, int k0) {
#pragma unroll
    for (int i = 0; i < 4; ++i) {
      int idx = t + i * 256;
      int r = idx >> 3, c = (idx & 7) << 3;          // 128 x 64 tiles
      async_b128(&As[buf][r][c], &A[(size_t)(m0 + r) * K + k0 + c]);
      async_b128(&Bs[buf][r][c], &Bt[(size_t)(n0 + r) * K + k0 + c]);
    }
  };

  const v8f zero = {0.f, 0.f, 0.f, 0.f, 0.f, 0.f, 0.f, 0.f};
  v8f acc[4][2];
#pragma unroll
  for (int i = 0; i < 4; ++i)
#pragma unroll
    for (int j = 0; j < 2; ++j) acc[i][j] = zero;

  stage(0, 0);
  wait_async0();
  __syncthreads();

  for (int k0 = 0; k0 < K; k0 += 64) {
    const int cur = (k0 >> 6) & 1;
    if (k0 + 64 < K) stage(cur ^ 1, k0 + 64);   // prefetch next tile via DMA

#pragma unroll
    for (int kk = 0; kk < 64; kk += 32) {
      v16h af[4], bf[2];
#pragma unroll
      for (int mt = 0; mt < 4; ++mt)
        af[mt] = load_a_frag(&As[cur][0][0], wm * 64 + mt * 16, 72, kk, lane);
#pragma unroll
      for (int nt = 0; nt < 2; ++nt)
        bf[nt] = load_bT_frag(&Bs[cur][0][0], wn * 32 + nt * 16, 72, kk, lane);
#pragma unroll
      for (int mt = 0; mt < 4; ++mt)
#pragma unroll
        for (int nt = 0; nt < 2; ++nt)
          acc[mt][nt] = __builtin_amdgcn_wmma_f32_16x16x32_f16(
              false, af[mt], false, bf[nt], (short)0, acc[mt][nt], false, false);
    }
    wait_async0();
    __syncthreads();
  }

  // epilogue: C layout -> VGPR r: row = base + (lane/16)*8 + r, col = lane&15
#pragma unroll
  for (int mt = 0; mt < 4; ++mt)
#pragma unroll
    for (int nt = 0; nt < 2; ++nt)
#pragma unroll
      for (int r = 0; r < 8; ++r) {
        int row = m0 + wm * 64 + mt * 16 + ((lane >> 4) << 3) + r;
        int col = n0 + wn * 32 + nt * 16 + (lane & 15);
        float val = acc[mt][nt][r] + bias[col];
        if (MODE == 0) {
          outf[(size_t)row * N + col] = val;
        } else {
          int b = row >> 11, s = row & (SEQ - 1);
          int h = col >> 6, dd = col & (HEAD_DIM - 1);
          if (MODE == 1)        // [B,H,S,d], pre-scaled (QSCALE for Q, 1 for K)
            outh[(((size_t)(b << 5) + h) * SEQ + s) * HEAD_DIM + dd] =
                (_Float16)(val * oscale);
          else                  // MODE 2: [B,H,d,S]
            outh[(((size_t)(b << 5) + h) * HEAD_DIM + dd) * SEQ + s] = (_Float16)val;
        }
      }
}

// ---------------------------------------------------------------------------
// Flash-attention style kernel. Q/K f16 in [B*H][S][64] (Q pre-scaled by
// QSCALE so scores are base-2 logits), V f16 in [B*H][64][S].
// Block: one (b,h) x 128 query rows; 8 waves, wave w owns rows [16w,16w+16).
// Double-buffered async staging of 64-key K/V chunks; online softmax in exp2
// domain; ctx kept in WMMA f32 accumulators (16x64 per wave); P re-staged
// through LDS for P@V.
// ---------------------------------------------------------------------------
__global__ __launch_bounds__(256) void attn_f16(const _Float16* __restrict__ Q,
                                                const _Float16* __restrict__ Km,
                                                const _Float16* __restrict__ Vt,
                                                _Float16* __restrict__ ctx) {
  __shared__ _Float16 Qs[128][72];         // 18 KB   [q][d]
  __shared__ _Float16 Ks[2][64][72];       // 18 KB   [key][d]
  __shared__ _Float16 Vts[2][64][72];      // 18 KB   [d][key]
  __shared__ _Float16 Ps[128][72];         // 18 KB   [q][key]
  const int t = threadIdx.x, lane = t & 31, w = t >> 5;
  const int bh = blockIdx.y;
  const int q0 = blockIdx.x * 128;
  const size_t base = (size_t)bh * SEQ * HEAD_DIM;

  auto stageKV = [&](int buf, int kb) {
#pragma unroll
    for (int i = 0; i < 2; ++i) {
      int idx = t + i * 256;
      int r = idx >> 3, c = (idx & 7) << 3;
      async_b128(&Ks[buf][r][c],  &Km[base + (size_t)(kb + r) * HEAD_DIM + c]);
      async_b128(&Vts[buf][r][c], &Vt[base + (size_t)r * SEQ + kb + c]);
    }
  };

  // async-stage Q block 128x64 once + first K/V chunk
#pragma unroll
  for (int i = 0; i < 4; ++i) {
    int idx = t + i * 256;
    int r = idx >> 3, c = (idx & 7) << 3;
    async_b128(&Qs[r][c], &Q[base + (size_t)(q0 + r) * HEAD_DIM + c]);
  }
  stageKV(0, 0);
  wait_async0();
  __syncthreads();

  const v8f zero = {0.f, 0.f, 0.f, 0.f, 0.f, 0.f, 0.f, 0.f};
  v8f o[4];
#pragma unroll
  for (int dt = 0; dt < 4; ++dt) o[dt] = zero;
  float m_run[8], l_run[8];
#pragma unroll
  for (int r = 0; r < 8; ++r) { m_run[r] = -3.0e38f; l_run[r] = 0.f; }

  for (int kb = 0; kb < SEQ; kb += 64) {
    const int cur = (kb >> 6) & 1;
    if (kb + 64 < SEQ) stageKV(cur ^ 1, kb + 64);   // DMA prefetch next chunk

    // scores (base-2 logits): S_w = Q_w (16x64) @ K_chunk^T -> 4 tiles 16x16
    v8f sc[4];
#pragma unroll
    for (int nt = 0; nt < 4; ++nt) sc[nt] = zero;
#pragma unroll
    for (int kk = 0; kk < 64; kk += 32) {
      v16h qa = load_a_frag(&Qs[0][0], w * 16, 72, kk, lane);
#pragma unroll
      for (int nt = 0; nt < 4; ++nt) {
        v16h kf = load_bT_frag(&Ks[cur][0][0], nt * 16, 72, kk, lane);
        sc[nt] = __builtin_amdgcn_wmma_f32_16x16x32_f16(
            false, qa, false, kf, (short)0, sc[nt], false, false);
      }
    }

    // online softmax in exp2 domain (per-row stats across 16-lane halves)
    float mx[8];
#pragma unroll
    for (int r = 0; r < 8; ++r)
      mx[r] = fmaxf(fmaxf(sc[0][r], sc[1][r]), fmaxf(sc[2][r], sc[3][r]));
#pragma unroll
    for (int off = 8; off >= 1; off >>= 1)
#pragma unroll
      for (int r = 0; r < 8; ++r) mx[r] = fmaxf(mx[r], __shfl_xor(mx[r], off, 32));

    float fac[8], ladd[8];
#pragma unroll
    for (int r = 0; r < 8; ++r) {
      float mn = fmaxf(m_run[r], mx[r]);
      fac[r] = exp2f(m_run[r] - mn);     // single v_exp_f32
      m_run[r] = mn;
      ladd[r] = 0.f;
    }
#pragma unroll
    for (int nt = 0; nt < 4; ++nt)
#pragma unroll
      for (int r = 0; r < 8; ++r) {
        float pv = exp2f(sc[nt][r] - m_run[r]);   // single v_exp_f32
        sc[nt][r] = pv;
        ladd[r] += pv;
      }
#pragma unroll
    for (int off = 8; off >= 1; off >>= 1)
#pragma unroll
      for (int r = 0; r < 8; ++r) ladd[r] += __shfl_xor(ladd[r], off, 32);
#pragma unroll
    for (int r = 0; r < 8; ++r) l_run[r] = l_run[r] * fac[r] + ladd[r];
#pragma unroll
    for (int dt = 0; dt < 4; ++dt)
#pragma unroll
      for (int r = 0; r < 8; ++r) o[dt][r] *= fac[r];

    // stage P (C-layout -> LDS -> A-layout); wave-private region, in-order DS
    {
      const int rr = (lane >> 4) << 3, cc = lane & 15;
#pragma unroll
      for (int nt = 0; nt < 4; ++nt)
#pragma unroll
        for (int r = 0; r < 8; ++r)
          Ps[w * 16 + rr + r][nt * 16 + cc] = (_Float16)sc[nt][r];
    }

    // ctx += P (16x64) @ V_chunk;  V fragment from transposed-V tile [d][key]
#pragma unroll
    for (int kk = 0; kk < 64; kk += 32) {
      v16h pa = load_a_frag(&Ps[0][0], w * 16, 72, kk, lane);
#pragma unroll
      for (int dt = 0; dt < 4; ++dt) {
        v16h vf = load_bT_frag(&Vts[cur][0][0], dt * 16, 72, kk, lane);
        o[dt] = __builtin_amdgcn_wmma_f32_16x16x32_f16(
            false, pa, false, vf, (short)0, o[dt], false, false);
      }
    }
    wait_async0();
    __syncthreads();
  }

  // normalize and store ctx as f16 in [B*S][HIDDEN] (merged heads) layout
  const int b = bh >> 5, h = bh & 31;
#pragma unroll
  for (int r = 0; r < 8; ++r) {
    const float inv = 1.0f / l_run[r];
    const int s = q0 + w * 16 + ((lane >> 4) << 3) + r;
#pragma unroll
    for (int dt = 0; dt < 4; ++dt) {
      int dd = dt * 16 + (lane & 15);
      ctx[((size_t)(b * SEQ + s)) * HIDDEN + h * HEAD_DIM + dd] =
          (_Float16)(o[dt][r] * inv);
    }
  }
}

// ---------------------------------------------------------------------------
// launcher
// ---------------------------------------------------------------------------
extern "C" void kernel_launch(void* const* d_in, const int* in_sizes, int n_in,
                              void* d_out, int out_size, void* d_ws, size_t ws_size,
                              hipStream_t stream) {
  const float* X  = (const float*)d_in[0];
  const float* Wq = (const float*)d_in[1];
  const float* bq = (const float*)d_in[2];
  const float* Wk = (const float*)d_in[3];
  const float* bk = (const float*)d_in[4];
  const float* Wv = (const float*)d_in[5];
  const float* bv = (const float*)d_in[6];
  const float* Wo = (const float*)d_in[7];
  const float* bo = (const float*)d_in[8];

  const size_t nX = (size_t)MROWS * HIDDEN;       // 8.39M
  const size_t nW = (size_t)HIDDEN * HIDDEN;      // 4.19M

  _Float16* p    = (_Float16*)d_ws;
  _Float16* Xh   = p; p += nX;
  _Float16* Wqt  = p; p += nW;   // transposed [N][K]
  _Float16* Wkt  = p; p += nW;
  _Float16* Wvt  = p; p += nW;
  _Float16* Wot  = p; p += nW;
  _Float16* Qh   = p; p += nX;   // [B*H][S][64]  (pre-scaled by QSCALE)
  _Float16* Kh   = p; p += nX;   // [B*H][S][64]
  _Float16* Vth  = p; p += nX;   // [B*H][64][S]
  _Float16* Ch   = p; p += nX;   // [B*S][HIDDEN]

  cvt_f32_f16<<<(int)((nX + 255) / 256), 256, 0, stream>>>(X, Xh, (int)nX);
  cvt_f32_f16_T<<<(int)((nW + 255) / 256), 256, 0, stream>>>(Wq, Wqt, HIDDEN, HIDDEN);
  cvt_f32_f16_T<<<(int)((nW + 255) / 256), 256, 0, stream>>>(Wk, Wkt, HIDDEN, HIDDEN);
  cvt_f32_f16_T<<<(int)((nW + 255) / 256), 256, 0, stream>>>(Wv, Wvt, HIDDEN, HIDDEN);
  cvt_f32_f16_T<<<(int)((nW + 255) / 256), 256, 0, stream>>>(Wo, Wot, HIDDEN, HIDDEN);

  dim3 gg(HIDDEN / 128, MROWS / 128);
  gemm_f16<1><<<gg, 256, 0, stream>>>(Xh, Wqt, bq, nullptr, Qh,  MROWS, HIDDEN, HIDDEN, QSCALE);
  gemm_f16<1><<<gg, 256, 0, stream>>>(Xh, Wkt, bk, nullptr, Kh,  MROWS, HIDDEN, HIDDEN, 1.0f);
  gemm_f16<2><<<gg, 256, 0, stream>>>(Xh, Wvt, bv, nullptr, Vth, MROWS, HIDDEN, HIDDEN, 1.0f);

  dim3 ga(SEQ / 128, BATCH * NUM_HEADS);
  attn_f16<<<ga, 256, 0, stream>>>(Qh, Kh, Vth, Ch);

  gemm_f16<0><<<gg, 256, 0, stream>>>(Ch, Wot, bo, (float*)d_out, nullptr, MROWS, HIDDEN, HIDDEN, 1.0f);
}
